// GAT_75299366633515
// MI455X (gfx1250) — matrix operand order repaired and checked by reading, not compiled, not run on hardware
//
#include <hip/hip_runtime.h>
#include <math.h>

// ---------------------------------------------------------------------------
// GAT (2-layer, PyG-style) for MI455X / gfx1250 (wave32, WMMA).
// GEMM1 (100000x1433 @ 1433x64) uses v_wmma_f32_16x16x32_bf16 with LDS tiling:
//  - branchless staging (row clamp + single guarded K-tail chunk)
//  - B staged transposed so A and B fragments are 2x ds_load_b128 each
//  - LDS stride 56 bf16 (112B = 16B*7) -> conflict-free b128 fragment reads
// Edge phases are memory/atomic bound: float atomics + gathers.
// ---------------------------------------------------------------------------

typedef __attribute__((ext_vector_type(16))) __bf16 v16bf;
typedef __attribute__((ext_vector_type(8)))  __bf16 v8bf;
typedef __attribute__((ext_vector_type(4)))  __bf16 v4bf;
typedef __attribute__((ext_vector_type(8)))  float  v8f;

#define NF1       1433   // layer-1 input features (K of GEMM1)
#define HF1       64     // H*F of layer 1
#define H1        8
#define F1        8
#define NCLS      7
#define NEG_SLOPE 0.2f
#define TILE_K    32
#define BLK_M     128
#define LDA       56     // bf16 elems per LDS row (112B = 16B * 7)
#define LDB       56

// ------------------------- staging: global f32 -> LDS bf16 -----------------
template<bool GUARD>
__device__ __forceinline__ void stage_tiles(const float* __restrict__ X,
                                            const float* __restrict__ W,
                                            __bf16* lA, __bf16* lBt,
                                            int rowBase, int k0, int N, int K,
                                            int tid) {
  // ---- A tile: 128 rows x 32 k. 512 tasks, 8 contiguous k each.
#pragma unroll
  for (int t = 0; t < 2; ++t) {
    int task = tid + t * 256;
    int r    = task >> 2;                 // 0..127
    int kk0  = (task & 3) << 3;           // 0,8,16,24
    size_t rowOff = (size_t)min(rowBase + r, N - 1) * K;
    v8bf pack;
#pragma unroll
    for (int j = 0; j < 8; ++j) {
      int gk = k0 + kk0 + j;
      float v;
      if (GUARD) {
        v = X[rowOff + min(gk, K - 1)];
        v = (gk < K) ? v : 0.0f;          // branchless select
      } else {
        v = X[rowOff + gk];
      }
      pack[j] = (__bf16)v;
    }
    *(v8bf*)&lA[r * LDA + kk0] = pack;    // ds_store_b128
  }
  // ---- B tile: 32 k x 64 n, stored TRANSPOSED: lBt[n][k]. 512 tasks, 4 k.
#pragma unroll
  for (int t = 0; t < 2; ++t) {
    int task = tid + t * 256;
    int c    = task & 63;                 // column (coalesced across lanes)
    int kk0  = (task >> 6) << 2;          // 0,4,...,28
    v4bf pack;
#pragma unroll
    for (int j = 0; j < 4; ++j) {
      int gk = k0 + kk0 + j;
      float v;
      if (GUARD) {
        v = W[(size_t)min(gk, K - 1) * HF1 + c];
        v = (gk < K) ? v : 0.0f;
      } else {
        v = W[(size_t)gk * HF1 + c];
      }
      pack[j] = (__bf16)v;
    }
    *(v4bf*)&lBt[c * LDB + kk0] = pack;   // ds_store_b64
  }
}

// ------------------------- one K-chunk of WMMAs ----------------------------
__device__ __forceinline__ void wmma_step(const __bf16* lA, const __bf16* lBt,
                                          int wave, int lane,
                                          v8f& a0, v8f& a1, v8f& a2, v8f& a3) {
  union F16x { v16bf v; v8bf h[2]; };
  // A frag (16x32): lane0-15 M=lane K=[0..7|16..23]; lane16-31 K=[8..15|24..31]
  const int arow = wave * 16 + (lane & 15);
  const int kb   = (lane >> 4) << 3;
  F16x af;
  af.h[0] = *(const v8bf*)&lA[arow * LDA + kb];        // ds_load_b128
  af.h[1] = *(const v8bf*)&lA[arow * LDA + 16 + kb];   // ds_load_b128
  // B frag (32x16): lane0-15 N=lane K=[0..7|16..23]; lane16-31 N=lane-16 K=[8..15|24..31]
  const int koff = (lane >> 4) << 3;
  const int ncol = lane & 15;
  F16x bf;
  bf.h[0] = *(const v8bf*)&lBt[ncol * LDB + koff];
  bf.h[1] = *(const v8bf*)&lBt[ncol * LDB + koff + 16];
  a0 = __builtin_amdgcn_wmma_f32_16x16x32_bf16(false, af.v, false, bf.v,
                                               (short)0, a0, false, false);
  bf.h[0] = *(const v8bf*)&lBt[(16 + ncol) * LDB + koff];
  bf.h[1] = *(const v8bf*)&lBt[(16 + ncol) * LDB + koff + 16];
  a1 = __builtin_amdgcn_wmma_f32_16x16x32_bf16(false, af.v, false, bf.v,
                                               (short)0, a1, false, false);
  bf.h[0] = *(const v8bf*)&lBt[(32 + ncol) * LDB + koff];
  bf.h[1] = *(const v8bf*)&lBt[(32 + ncol) * LDB + koff + 16];
  a2 = __builtin_amdgcn_wmma_f32_16x16x32_bf16(false, af.v, false, bf.v,
                                               (short)0, a2, false, false);
  bf.h[0] = *(const v8bf*)&lBt[(48 + ncol) * LDB + koff];
  bf.h[1] = *(const v8bf*)&lBt[(48 + ncol) * LDB + koff + 16];
  a3 = __builtin_amdgcn_wmma_f32_16x16x32_bf16(false, af.v, false, bf.v,
                                               (short)0, a3, false, false);
}

// ------------------------- GEMM1: h1 = X @ W1 (bf16 WMMA) ------------------
__global__ __launch_bounds__(256)
void gat1_gemm_wmma(const float* __restrict__ X, const float* __restrict__ W,
                    float* __restrict__ h1, int N, int K) {
  __shared__ __bf16 lA[BLK_M * LDA];
  __shared__ __bf16 lBt[HF1 * LDB];

  const int tid  = threadIdx.x;
  const int lane = tid & 31;
  const int wave = tid >> 5;                 // 0..7 -> 16-row strip
  const int rowBase = blockIdx.x * BLK_M;

  v8f a0 = {}, a1 = {}, a2 = {}, a3 = {};

  const int Kmain = (K / TILE_K) * TILE_K;
  for (int k0 = 0; k0 < Kmain; k0 += TILE_K) {
    stage_tiles<false>(X, W, lA, lBt, rowBase, k0, N, K, tid);
    // prefetch next A chunk (global_prefetch_b8)
    if (k0 + TILE_K < K) {
      int pr = min(rowBase + (tid >> 1), N - 1);
      __builtin_prefetch(&X[(size_t)pr * K + k0 + TILE_K + ((tid & 1) << 4)], 0, 0);
    }
    __syncthreads();
    wmma_step(lA, lBt, wave, lane, a0, a1, a2, a3);
    __syncthreads();
  }
  if (Kmain < K) {
    stage_tiles<true>(X, W, lA, lBt, rowBase, Kmain, N, K, tid);
    __syncthreads();
    wmma_step(lA, lBt, wave, lane, a0, a1, a2, a3);
  }

  // ---- store: C VGPR r -> M = r (+8 for high lanes), N = lane&15
  const int m0   = rowBase + wave * 16 + ((lane >> 4) << 3);
  const int ncol = lane & 15;
#pragma unroll
  for (int r = 0; r < 8; ++r) {
    int gm = m0 + r;
    if (gm < N) {
      float* o = h1 + (size_t)gm * HF1;
      o[ncol]      = a0[r];
      o[16 + ncol] = a1[r];
      o[32 + ncol] = a2[r];
      o[48 + ncol] = a3[r];
    }
  }
}

// ------------------------- generic fill ------------------------------------
__global__ void fill_f32(float* __restrict__ p, float v, size_t n) {
  size_t i = (size_t)blockIdx.x * blockDim.x + threadIdx.x;
  if (i < n) p[i] = v;
}

// ------------------------- per-node attention coefficients (layer 1) -------
__global__ void node_logits1(const float* __restrict__ h1,
                             const float* __restrict__ attS,
                             const float* __restrict__ attD,
                             float* __restrict__ aS, float* __restrict__ aD,
                             int N) {
  int i = blockIdx.x * blockDim.x + threadIdx.x;   // (n, h)
  if (i >= N * H1) return;
  int n = i >> 3, h = i & 7;
  const float* hp = h1 + (size_t)n * HF1 + h * F1;
  float s = 0.0f, d = 0.0f;
#pragma unroll
  for (int f = 0; f < F1; ++f) {
    s += hp[f] * attS[h * F1 + f];
    d += hp[f] * attD[h * F1 + f];
  }
  aS[i] = s;
  aD[i] = d;
}

// ------------------------- helpers -----------------------------------------
__device__ __forceinline__ void atomicMaxF(float* addr, float val) {
  unsigned* ua = (unsigned*)addr;
  unsigned old = __hip_atomic_load(ua, __ATOMIC_RELAXED, __HIP_MEMORY_SCOPE_AGENT);
  while (__uint_as_float(old) < val) {
    unsigned assumed = old;
    old = atomicCAS(ua, assumed, __float_as_uint(val));
    if (old == assumed) break;
  }
}

__device__ __forceinline__ float lrelu(float e) {
  return e > 0.0f ? e : NEG_SLOPE * e;
}

// ------------------------- layer-1 edge passes (H=8) -----------------------
__global__ void edge1_max(const long long* __restrict__ src,
                          const long long* __restrict__ dst,
                          long long E, int N,
                          const float* __restrict__ aS,
                          const float* __restrict__ aD,
                          float* __restrict__ m) {
  long long idx = (long long)blockIdx.x * blockDim.x + threadIdx.x;
  long long tot = ((long long)E + N) * H1;
  if (idx >= tot) return;
  long long e = idx >> 3; int h = (int)(idx & 7);
  long long s = (e < E) ? src[e] : (e - E);
  long long d = (e < E) ? dst[e] : (e - E);
  float lg = lrelu(aS[s * H1 + h] + aD[d * H1 + h]);
  atomicMaxF(&m[d * H1 + h], lg);
}

__global__ void edge1_sum(const long long* __restrict__ src,
                          const long long* __restrict__ dst,
                          long long E, int N,
                          const float* __restrict__ aS,
                          const float* __restrict__ aD,
                          const float* __restrict__ m,
                          float* __restrict__ ssum) {
  long long idx = (long long)blockIdx.x * blockDim.x + threadIdx.x;
  long long tot = ((long long)E + N) * H1;
  if (idx >= tot) return;
  long long e = idx >> 3; int h = (int)(idx & 7);
  long long s = (e < E) ? src[e] : (e - E);
  long long d = (e < E) ? dst[e] : (e - E);
  float lg = lrelu(aS[s * H1 + h] + aD[d * H1 + h]);
  atomicAdd(&ssum[d * H1 + h], __expf(lg - m[d * H1 + h]));
}

__global__ void edge1_agg(const long long* __restrict__ src,
                          const long long* __restrict__ dst,
                          long long E, int N,
                          const float* __restrict__ aS,
                          const float* __restrict__ aD,
                          const float* __restrict__ m,
                          const float* __restrict__ ssum,
                          const float* __restrict__ h1,
                          float* __restrict__ agg) {
  long long idx = (long long)blockIdx.x * blockDim.x + threadIdx.x;
  long long tot = ((long long)E + N) * H1;
  if (idx >= tot) return;
  long long e = idx >> 3; int h = (int)(idx & 7);
  long long s = (e < E) ? src[e] : (e - E);
  long long d = (e < E) ? dst[e] : (e - E);
  float lg = lrelu(aS[s * H1 + h] + aD[d * H1 + h]);
  float alpha = __expf(lg - m[d * H1 + h]) / ssum[d * H1 + h];
  const float* hs = h1 + (size_t)s * HF1 + h * F1;
  float* o = agg + (size_t)d * HF1 + h * F1;
#pragma unroll
  for (int f = 0; f < F1; ++f) atomicAdd(&o[f], alpha * hs[f]);
}

// ------------------------- bias + ELU (in place) ---------------------------
__global__ void bias_elu(float* __restrict__ a, const float* __restrict__ b,
                         size_t n) {
  size_t i = (size_t)blockIdx.x * blockDim.x + threadIdx.x;
  if (i >= n) return;
  float v = a[i] + b[i % HF1];
  a[i] = v > 0.0f ? v : expm1f(v);
}

// ------------------------- layer 2: GEMM + node logits ---------------------
__global__ void gat2_gemm(const float* __restrict__ hx,
                          const float* __restrict__ W2,
                          const float* __restrict__ attS,
                          const float* __restrict__ attD,
                          float* __restrict__ h2,
                          float* __restrict__ aS, float* __restrict__ aD,
                          int N) {
  int n = blockIdx.x * blockDim.x + threadIdx.x;
  if (n >= N) return;
  float hreg[HF1];
  const float* hp = hx + (size_t)n * HF1;
#pragma unroll
  for (int k = 0; k < HF1; ++k) hreg[k] = hp[k];
  float s = 0.0f, d = 0.0f;
#pragma unroll
  for (int c = 0; c < NCLS; ++c) {
    float acc = 0.0f;
#pragma unroll
    for (int k = 0; k < HF1; ++k) acc += hreg[k] * W2[k * NCLS + c];
    h2[(size_t)n * NCLS + c] = acc;
    s += acc * attS[c];
    d += acc * attD[c];
  }
  aS[n] = s;
  aD[n] = d;
}

// ------------------------- layer-2 edge passes (H=1) -----------------------
__global__ void edge2_max(const long long* __restrict__ src,
                          const long long* __restrict__ dst,
                          long long E, int N,
                          const float* __restrict__ aS,
                          const float* __restrict__ aD,
                          float* __restrict__ m) {
  long long e = (long long)blockIdx.x * blockDim.x + threadIdx.x;
  if (e >= E + N) return;
  long long s = (e < E) ? src[e] : (e - E);
  long long d = (e < E) ? dst[e] : (e - E);
  atomicMaxF(&m[d], lrelu(aS[s] + aD[d]));
}

__global__ void edge2_sum(const long long* __restrict__ src,
                          const long long* __restrict__ dst,
                          long long E, int N,
                          const float* __restrict__ aS,
                          const float* __restrict__ aD,
                          const float* __restrict__ m,
                          float* __restrict__ ssum) {
  long long e = (long long)blockIdx.x * blockDim.x + threadIdx.x;
  if (e >= E + N) return;
  long long s = (e < E) ? src[e] : (e - E);
  long long d = (e < E) ? dst[e] : (e - E);
  atomicAdd(&ssum[d], __expf(lrelu(aS[s] + aD[d]) - m[d]));
}

__global__ void edge2_agg(const long long* __restrict__ src,
                          const long long* __restrict__ dst,
                          long long E, int N,
                          const float* __restrict__ aS,
                          const float* __restrict__ aD,
                          const float* __restrict__ m,
                          const float* __restrict__ ssum,
                          const float* __restrict__ h2,
                          float* __restrict__ agg) {
  long long e = (long long)blockIdx.x * blockDim.x + threadIdx.x;
  if (e >= E + N) return;
  long long s = (e < E) ? src[e] : (e - E);
  long long d = (e < E) ? dst[e] : (e - E);
  float alpha = __expf(lrelu(aS[s] + aD[d]) - m[d]) / ssum[d];
  const float* hs = h2 + (size_t)s * NCLS;
  float* o = agg + (size_t)d * NCLS;
#pragma unroll
  for (int c = 0; c < NCLS; ++c) atomicAdd(&o[c], alpha * hs[c]);
}

// ------------------------- bias + log_softmax ------------------------------
__global__ void final_logsoftmax(const float* __restrict__ agg,
                                 const float* __restrict__ b2,
                                 float* __restrict__ out, int N) {
  int n = blockIdx.x * blockDim.x + threadIdx.x;
  if (n >= N) return;
  float v[NCLS];
  float mx = -INFINITY;
#pragma unroll
  for (int c = 0; c < NCLS; ++c) {
    v[c] = agg[(size_t)n * NCLS + c] + b2[c];
    mx = fmaxf(mx, v[c]);
  }
  float sum = 0.0f;
#pragma unroll
  for (int c = 0; c < NCLS; ++c) sum += __expf(v[c] - mx);
  float lse = mx + __logf(sum);
#pragma unroll
  for (int c = 0; c < NCLS; ++c) out[(size_t)n * NCLS + c] = v[c] - lse;
}

// ---------------------------------------------------------------------------
extern "C" void kernel_launch(void* const* d_in, const int* in_sizes, int n_in,
                              void* d_out, int out_size, void* d_ws, size_t ws_size,
                              hipStream_t stream) {
  (void)n_in; (void)out_size; (void)ws_size;
  const float*     X     = (const float*)d_in[0];
  const long long* eidx  = (const long long*)d_in[1];   // int64 [2, E]
  const float*     W1    = (const float*)d_in[2];
  const float*     attS1 = (const float*)d_in[3];
  const float*     attD1 = (const float*)d_in[4];
  const float*     b1    = (const float*)d_in[5];
  const float*     W2    = (const float*)d_in[6];
  const float*     attS2 = (const float*)d_in[7];
  const float*     attD2 = (const float*)d_in[8];
  const float*     b2    = (const float*)d_in[9];

  const int       N    = in_sizes[0] / NF1;
  const long long E    = in_sizes[1] / 2;
  const long long Etot = E + N;
  const long long* esrc = eidx;
  const long long* edst = eidx + E;

  // ---- workspace carve (floats) ----
  float* p    = (float*)d_ws;
  float* h1   = p; p += (size_t)N * HF1;   // layer-1 transformed features
  float* aS1  = p; p += (size_t)N * H1;
  float* aD1  = p; p += (size_t)N * H1;
  float* m1   = p; p += (size_t)N * H1;
  float* s1   = p; p += (size_t)N * H1;
  float* agg1 = p; p += (size_t)N * HF1;   // -> becomes ELU'd layer-2 input
  float* h2   = p; p += (size_t)N * NCLS;
  float* aS2  = p; p += (size_t)N;
  float* aD2  = p; p += (size_t)N;
  float* m2   = p; p += (size_t)N;
  float* s2   = p; p += (size_t)N;
  float* agg2 = p; p += (size_t)N * NCLS;

  const int TB = 256;
  auto blocks = [](long long n, int tb) { return (unsigned)((n + tb - 1) / tb); };

  // ---- layer 1 ----
  gat1_gemm_wmma<<<(N + BLK_M - 1) / BLK_M, 256, 0, stream>>>(X, W1, h1, N, NF1);
  node_logits1<<<blocks((long long)N * H1, TB), TB, 0, stream>>>(h1, attS1, attD1,
                                                                 aS1, aD1, N);
  fill_f32<<<blocks((long long)N * H1, TB), TB, 0, stream>>>(m1, -INFINITY,
                                                             (size_t)N * H1);
  fill_f32<<<blocks((long long)N * H1, TB), TB, 0, stream>>>(s1, 0.0f,
                                                             (size_t)N * H1);
  fill_f32<<<blocks((long long)N * HF1, TB), TB, 0, stream>>>(agg1, 0.0f,
                                                              (size_t)N * HF1);
  edge1_max<<<blocks(Etot * H1, TB), TB, 0, stream>>>(esrc, edst, E, N, aS1, aD1, m1);
  edge1_sum<<<blocks(Etot * H1, TB), TB, 0, stream>>>(esrc, edst, E, N, aS1, aD1,
                                                      m1, s1);
  edge1_agg<<<blocks(Etot * H1, TB), TB, 0, stream>>>(esrc, edst, E, N, aS1, aD1,
                                                      m1, s1, h1, agg1);
  bias_elu<<<blocks((long long)N * HF1, TB), TB, 0, stream>>>(agg1, b1,
                                                              (size_t)N * HF1);

  // ---- layer 2 ----
  gat2_gemm<<<blocks(N, TB), TB, 0, stream>>>(agg1, W2, attS2, attD2, h2, aS2, aD2, N);
  fill_f32<<<blocks(N, TB), TB, 0, stream>>>(m2, -INFINITY, (size_t)N);
  fill_f32<<<blocks(N, TB), TB, 0, stream>>>(s2, 0.0f, (size_t)N);
  fill_f32<<<blocks((long long)N * NCLS, TB), TB, 0, stream>>>(agg2, 0.0f,
                                                               (size_t)N * NCLS);
  edge2_max<<<blocks(Etot, TB), TB, 0, stream>>>(esrc, edst, E, N, aS2, aD2, m2);
  edge2_sum<<<blocks(Etot, TB), TB, 0, stream>>>(esrc, edst, E, N, aS2, aD2, m2, s2);
  edge2_agg<<<blocks(Etot, TB), TB, 0, stream>>>(esrc, edst, E, N, aS2, aD2, m2, s2,
                                                 h2, agg2);
  final_logsoftmax<<<blocks(N, TB), TB, 0, stream>>>(agg2, b2, (float*)d_out, N);
}
